// CustomClassifier_56307021251047
// MI455X (gfx1250) — matrix operand order repaired and checked by reading, not compiled, not run on hardware
//
#include <hip/hip_runtime.h>

#define F_IN      2048
#define D_SK      8192
#define LOGD      13
#define N_CLASSES 14951
#define ROW_STRIDE 40   // bf16 elems per LDS A row: 32 data + 8 pad (80B stride kills bank conflicts)

typedef __attribute__((ext_vector_type(16))) __bf16 v16bf;
typedef __attribute__((ext_vector_type(8)))  __bf16 v8bf;
typedef __attribute__((ext_vector_type(8)))  float  v8f;

// --- CDNA5 async global->LDS copy (ASYNCcnt-tracked, no VGPR data) ----------
__device__ __forceinline__ void async_lds_b128(unsigned lds_off, const void* gptr) {
  asm volatile("global_load_async_to_lds_b128 %0, %1, off"
               :: "v"(lds_off), "v"(gptr) : "memory");
}
__device__ __forceinline__ void wait_async0() {
  asm volatile("s_wait_asynccnt 0x0" ::: "memory");
}

// ---------------------------------------------------------------------------
// In-place radix-2 DIT complex FFT, N=8192, 256 threads, data in LDS.
// ---------------------------------------------------------------------------
__device__ __forceinline__ void fft8192(float2* Z, int t) {
  for (int j = t; j < D_SK; j += 256) {
    int r = (int)(__brev((unsigned)j) >> (32 - LOGD));
    if (r > j) { float2 tmp = Z[j]; Z[j] = Z[r]; Z[r] = tmp; }
  }
  __syncthreads();
  int s = 0;
  for (int m = 1; m < D_SK; m <<= 1, ++s) {
    float angb = -3.14159265358979323846f / (float)m;
    for (int idx = t; idx < (D_SK >> 1); idx += 256) {
      int j  = idx & (m - 1);
      int i0 = ((idx >> s) << (s + 1)) + j;
      int i1 = i0 + m;
      float sa, ca;
      __sincosf(angb * (float)j, &sa, &ca);
      float2 a = Z[i0];
      float2 b = Z[i1];
      float tr = ca * b.x - sa * b.y;
      float ti = ca * b.y + sa * b.x;
      Z[i0] = make_float2(a.x + tr, a.y + ti);
      Z[i1] = make_float2(a.x - tr, a.y - ti);
    }
    __syncthreads();
  }
}

// ---------------------------------------------------------------------------
// Kernel 1: fused count-sketch + circular convolution (packed complex FFT).
// ---------------------------------------------------------------------------
__global__ __launch_bounds__(256) void cbp_sketch_fft(
    const float* __restrict__ x,
    const int*   __restrict__ h1, const int*   __restrict__ h2,
    const float* __restrict__ s1, const float* __restrict__ s2,
    __bf16*      __restrict__ cbp)
{
  __shared__ float2 Z[D_SK];           // 64 KB
  const int t = threadIdx.x;
  const int b = blockIdx.x;

  for (int j = t; j < D_SK; j += 256) Z[j] = make_float2(0.f, 0.f);
  __syncthreads();

  const float* xr = x + (size_t)b * F_IN;
  for (int i = t; i < F_IN; i += 256) {
    float xv = xr[i];
    atomicAdd(&Z[h1[i]].x, s1[i] * xv);   // sk1 -> real part
    atomicAdd(&Z[h2[i]].y, s2[i] * xv);   // sk2 -> imag part
  }
  __syncthreads();

  fft8192(Z, t);   // Z[f] = DFT(sk1 + i*sk2)

  if (t == 0) {
    float2 z0 = Z[0];
    Z[0] = make_float2(z0.x * z0.y, 0.f);
    float2 zh = Z[D_SK / 2];
    Z[D_SK / 2] = make_float2(zh.x * zh.y, 0.f);
  }
  for (int f = 1 + t; f < D_SK / 2; f += 256) {
    float2 zf = Z[f];
    float2 zc = Z[D_SK - f];
    float s1x = 0.5f * (zf.x + zc.x);
    float s1y = 0.5f * (zf.y - zc.y);
    float s2x = 0.5f * (zf.y + zc.y);
    float s2y = 0.5f * (zc.x - zf.x);
    float px = s1x * s2x - s1y * s2y;
    float py = s1x * s2y + s1y * s2x;
    Z[f]        = make_float2(px, -py);   // conj(P)[f]
    Z[D_SK - f] = make_float2(px,  py);   // conj(P)[N-f] = P[f]
  }
  __syncthreads();

  fft8192(Z, t);   // FFT(conj(P)); ifft(P) = conj(.)/N -> take Re(.)/N

  const float inv = 1.0f / (float)D_SK;
  __bf16* outr = cbp + (size_t)b * D_SK;
  for (int k = t; k < D_SK; k += 256)
    outr[k] = (__bf16)(Z[k].x * inv);
}

// ---------------------------------------------------------------------------
// Kernel 2: out = cbp_bf16 @ W^T + b via v_wmma_f32_16x16x32_bf16.
// 8 waves/block, each owns 16 classes x all 256 batch rows (W read once).
// A: async global->LDS, double-buffered; A fragments rotated 3-deep so
// ds_load latency hides under WMMA. B (W): f32 loads pipelined one full
// K-iteration ahead, converted to bf16 in registers.
// ---------------------------------------------------------------------------
__global__ __launch_bounds__(256) void cbp_gemm_wmma(
    const __bf16* __restrict__ A,     // [256][D_SK] bf16 (cbp)
    const float*  __restrict__ W,     // [N_CLASSES][D_SK] f32
    const float*  __restrict__ bias,  // [N_CLASSES]
    float*        __restrict__ out)   // [256][N_CLASSES]
{
  __shared__ __bf16 ldsA[2][256 * ROW_STRIDE];   // 2 x 20 KB

  const int t     = threadIdx.x;
  const int lane  = t & 31;
  const int wave  = t >> 5;
  const int lcol  = lane & 15;
  const int lhalf = lane >> 4;
  const int n     = blockIdx.x * 128 + wave * 16 + lcol;
  const bool nval = (n < N_CLASSES);

  v8f acc[16];
  const v8f vz = {0.f, 0.f, 0.f, 0.f, 0.f, 0.f, 0.f, 0.f};
#pragma unroll
  for (int m = 0; m < 16; ++m) acc[m] = vz;

  // async-stage K-chunk k0 into buffer buf: thread t copies row t's 64 bytes
  auto stage = [&](int buf, int k0) {
    const char* g = (const char*)(A + (size_t)t * D_SK + k0);
    unsigned l = (unsigned)(size_t)(&ldsA[buf][t * ROW_STRIDE]);
    async_lds_b128(l,      g);
    async_lds_b128(l + 16, g + 16);
    async_lds_b128(l + 32, g + 32);
    async_lds_b128(l + 48, g + 48);
  };

  // A fragment (16x32 bf16) per ISA lane layout:
  // lanes 0-15: K{0..7,16..23}; lanes 16-31: K{8..15,24..31}; row M = lcol.
  auto load_frag = [&](int buf, int m) {
    const __bf16* base = &ldsA[buf][(m * 16 + lcol) * ROW_STRIDE];
    v8bf lo = *(const v8bf*)(base + lhalf * 8);
    v8bf hi = *(const v8bf*)(base + 16 + lhalf * 8);
    v16bf a;
#pragma unroll
    for (int e = 0; e < 8; ++e) { a[e] = lo[e]; a[8 + e] = hi[e]; }
    return a;
  };

  // B pipeline registers: W f32 for the *current* K-chunk, loaded one
  // iteration in advance.
  float4 wf0, wf1, wf2, wf3;
  if (nval) {
    const float4* wp = (const float4*)(W + (size_t)n * D_SK + lhalf * 8);
    wf0 = wp[0]; wf1 = wp[1]; wf2 = wp[4]; wf3 = wp[5];
  }

  stage(0, 0);
  wait_async0();
  __syncthreads();

  int p = 0;
#pragma unroll 1
  for (int k0 = 0; k0 < D_SK; k0 += 32) {
    if (k0 + 32 < D_SK) stage(p ^ 1, k0 + 32);   // async, overlaps compute

    // Convert the prefetched W registers -> bf16 B fragment, then
    // immediately issue next iteration's W loads (full K-step of latency).
    v16bf bf;
    if (nval) {
      bf[0]  = (__bf16)wf0.x; bf[1]  = (__bf16)wf0.y; bf[2]  = (__bf16)wf0.z; bf[3]  = (__bf16)wf0.w;
      bf[4]  = (__bf16)wf1.x; bf[5]  = (__bf16)wf1.y; bf[6]  = (__bf16)wf1.z; bf[7]  = (__bf16)wf1.w;
      bf[8]  = (__bf16)wf2.x; bf[9]  = (__bf16)wf2.y; bf[10] = (__bf16)wf2.z; bf[11] = (__bf16)wf2.w;
      bf[12] = (__bf16)wf3.x; bf[13] = (__bf16)wf3.y; bf[14] = (__bf16)wf3.z; bf[15] = (__bf16)wf3.w;
      if (k0 + 32 < D_SK) {
        const float4* wp =
            (const float4*)(W + (size_t)n * D_SK + (k0 + 32) + lhalf * 8);
        wf0 = wp[0]; wf1 = wp[1]; wf2 = wp[4]; wf3 = wp[5];
      }
    } else {
#pragma unroll
      for (int e = 0; e < 16; ++e) bf[e] = (__bf16)0.f;
    }

    // 3-deep rotated A fragments: fragment m+2's ds_loads issue before
    // wmma m consumes fragment m -> waitcnt becomes dscnt<=4, not 0.
    v16bf afr[3];
    afr[0] = load_frag(p, 0);
    afr[1] = load_frag(p, 1);
#pragma unroll
    for (int m = 0; m < 16; ++m) {
      if (m + 2 < 16) afr[(m + 2) % 3] = load_frag(p, m + 2);
      acc[m] = __builtin_amdgcn_wmma_f32_16x16x32_bf16(
          false, afr[m % 3], false, bf, (short)0, acc[m], false, false);
    }

    wait_async0();     // next buffer's async copies landed (per-wave)
    __syncthreads();   // publish across waves
    p ^= 1;
  }

  if (nval) {
    float bv = bias[n];
#pragma unroll
    for (int m = 0; m < 16; ++m) {
#pragma unroll
      for (int r = 0; r < 8; ++r) {
        int row = m * 16 + lhalf * 8 + r;   // C/D layout: lanes 16-31 are M+8
        out[(size_t)row * N_CLASSES + n] = acc[m][r] + bv;
      }
    }
  }
}

// ---------------------------------------------------------------------------
extern "C" void kernel_launch(void* const* d_in, const int* in_sizes, int n_in,
                              void* d_out, int out_size, void* d_ws, size_t ws_size,
                              hipStream_t stream) {
  const float* x    = (const float*)d_in[0];
  const int*   h1   = (const int*)  d_in[1];
  const int*   h2   = (const int*)  d_in[2];
  const float* s1   = (const float*)d_in[3];
  const float* s2   = (const float*)d_in[4];
  const float* W    = (const float*)d_in[5];
  const float* bias = (const float*)d_in[6];
  float*  out = (float*)d_out;
  __bf16* cbp = (__bf16*)d_ws;   // 256*8192 bf16 = 4 MB

  cbp_sketch_fft<<<256, 256, 0, stream>>>(x, h1, h2, s1, s2, cbp);

  const int nblocks = (N_CLASSES + 127) / 128;  // 117
  cbp_gemm_wmma<<<nblocks, 256, 0, stream>>>(cbp, W, bias, out);
}
